// CrossAttention2D_77713138254430
// MI455X (gfx1250) — compile-verified, matrix operand order checked by reading, not run on hardware
//
#include <hip/hip_runtime.h>

// ---------- types ----------
typedef _Float16 v16h __attribute__((ext_vector_type(16)));
typedef _Float16 v8h  __attribute__((ext_vector_type(8)));
typedef _Float16 v4h  __attribute__((ext_vector_type(4)));
typedef float    v8f  __attribute__((ext_vector_type(8)));
typedef unsigned int v4u __attribute__((ext_vector_type(4)));
typedef unsigned int v8u __attribute__((ext_vector_type(8)));

static __device__ __forceinline__ v8f zf() {
    v8f z;
#pragma unroll
    for (int i = 0; i < 8; ++i) z[i] = 0.0f;
    return z;
}

static __device__ __forceinline__ v8f wmma16(v16h a, v16h b, v8f c) {
    return __builtin_amdgcn_wmma_f32_16x16x32_f16(
        false, a, false, b, (short)0, c, false, false);
}

// ---------- DPP lane-reduction within 16-lane halves (VALU only) ----------
template <int CTRL>
static __device__ __forceinline__ float dppf(float x) {
    int r = __builtin_amdgcn_update_dpp(0, __float_as_int(x), CTRL, 0xf, 0xf, true);
    return __int_as_float(r);
}
static __device__ __forceinline__ float red_max16(float x) {
    x = fmaxf(x, dppf<0xB1>(x));    // quad_perm(1,0,3,2)
    x = fmaxf(x, dppf<0x4E>(x));    // quad_perm(2,3,0,1)
    x = fmaxf(x, dppf<0x141>(x));   // row_half_mirror
    x = fmaxf(x, dppf<0x140>(x));   // row_mirror
    return x;
}
static __device__ __forceinline__ float red_add16(float x) {
    x += dppf<0xB1>(x);
    x += dppf<0x4E>(x);
    x += dppf<0x141>(x);
    x += dppf<0x140>(x);
    return x;
}

// ---------- fragment load (ISA 7.12.2 16-bit A layout) ----------
static __device__ __forceinline__ v16h load_frag(const _Float16* base, int row,
                                                 int stride, int koff, int lane) {
    const _Float16* p = base + (size_t)(row + (lane & 15)) * stride + koff + ((lane >> 4) << 3);
    union { v16h v; v8h h[2]; } u;
    u.h[0] = *(const v8h*)(p);
    u.h[1] = *(const v8h*)(p + 16);
    return u.v;
}

static __device__ __forceinline__ v4h ld4h(const float* p) {
    const float4 f = *(const float4*)p;
    v4h r; r[0] = (_Float16)f.x; r[1] = (_Float16)f.y; r[2] = (_Float16)f.z; r[3] = (_Float16)f.w;
    return r;
}
static __device__ __forceinline__ v4h ld4h(const _Float16* p) {
    return *(const v4h*)p;
}

// ---------- Tensor Data Mover: 2D f16 64x64 tile -> LDS (stride 72 halves) --
static __device__ __forceinline__ void tdm_load_k64x64(unsigned int lds_off,
                                                       const _Float16* gptr) {
    unsigned long long ga = (unsigned long long)(uintptr_t)gptr;
    v4u g0;
    g0[0] = 1u;                                    // count=1 (user descriptor)
    g0[1] = lds_off;                               // lds_addr
    g0[2] = (unsigned int)ga;                      // global_addr[31:0]
    g0[3] = (unsigned int)(ga >> 32) | (2u << 30); // global_addr[56:32] | type=2
    v8u g1;
    g1[0] = (1u << 16) | (1u << 20) | (4u << 22) | (3u << 25);
    //       data_size=2B | pad_en | pad_interval=32dw(128B) | pad_amount=4dw(16B)
    g1[1] = (64u << 16);   // tensor_dim0 lo16 = 64
    g1[2] = (64u << 16);   // tensor_dim0 hi=0 | tensor_dim1 lo16 = 64
    g1[3] = (64u << 16);   // tensor_dim1 hi=0 | tile_dim0 = 64
    g1[4] = 64u;           // tile_dim1 = 64, tile_dim2 = 0
    g1[5] = 64u;           // tensor_dim0_stride = 64 elements
    g1[6] = 0u;
    g1[7] = 0u;
    asm volatile("tensor_load_to_lds %0, %1" :: "s"(g0), "s"(g1) : "memory");
}
static __device__ __forceinline__ void wait_tensorcnt0() {
#if __has_builtin(__builtin_amdgcn_s_wait_tensorcnt)
    __builtin_amdgcn_s_wait_tensorcnt(0);
#else
    asm volatile("s_wait_tensorcnt 0x0" ::: "memory");
#endif
}

// ---------- GEMM + bias:  C[M,N] = A[M,K] * W[K,N] + bias ----------
// 128x128x32 block tile, double-buffered LDS, register-staged global loads:
// one barrier per K-step; global loads for step i+1 overlap WMMAs of step i.
#define GB_LDA 40   // f16 row stride in LDS (80B: 16B-aligned, non-pow2)

template <typename AT, bool HEADSPLIT, typename OT>
__global__ __launch_bounds__(256) void gemm_bias_kernel(
    const AT* __restrict__ A, const float* __restrict__ W,
    const float* __restrict__ bias, OT* __restrict__ out,
    int M, int Kd, int N, int S)
{
    __shared__ alignas(16) _Float16 As[2][128][GB_LDA];
    __shared__ alignas(16) _Float16 Bs[2][128][GB_LDA];   // Bs[n][k]

    const int tid  = threadIdx.x;
    const int lane = tid & 31;
    const int w    = tid >> 5;
    const int wm   = w & 3;    // 32-row slab
    const int wn   = w >> 2;   // 64-col slab
    const int r0   = blockIdx.y * 128;
    const int c0   = blockIdx.x * 128;

    // per-thread staging coordinates
    int aRow[4], aKq[4], wKk[4], wNq[4];
#pragma unroll
    for (int t = 0; t < 4; ++t) {
        int idx = tid + t * 256;
        aRow[t] = idx >> 3;          aKq[t] = (idx & 7) << 2;
        wKk[t]  = idx >> 5;          wNq[t] = (idx & 31) << 2;
    }

    v4h   aReg[4];
    float4 wReg[4];
    auto gload = [&](int kb) {
#pragma unroll
        for (int t = 0; t < 4; ++t)
            aReg[t] = ld4h(A + (size_t)(r0 + aRow[t]) * Kd + kb + aKq[t]);
#pragma unroll
        for (int t = 0; t < 4; ++t)
            wReg[t] = *(const float4*)(W + (size_t)(kb + wKk[t]) * N + c0 + wNq[t]);
    };
    auto lstore = [&](int buf) {
#pragma unroll
        for (int t = 0; t < 4; ++t)
            *(v4h*)&As[buf][aRow[t]][aKq[t]] = aReg[t];
#pragma unroll
        for (int t = 0; t < 4; ++t) {
            Bs[buf][wNq[t] + 0][wKk[t]] = (_Float16)wReg[t].x;
            Bs[buf][wNq[t] + 1][wKk[t]] = (_Float16)wReg[t].y;
            Bs[buf][wNq[t] + 2][wKk[t]] = (_Float16)wReg[t].z;
            Bs[buf][wNq[t] + 3][wKk[t]] = (_Float16)wReg[t].w;
        }
    };

    v8f acc[2][4];
#pragma unroll
    for (int mt = 0; mt < 2; ++mt)
#pragma unroll
        for (int nt = 0; nt < 4; ++nt) acc[mt][nt] = zf();

    gload(0);
    lstore(0);
    __syncthreads();

    int buf = 0;
    for (int kb = 0; kb < Kd; kb += 32, buf ^= 1) {
        const bool hasNext = (kb + 32 < Kd);
        if (hasNext) gload(kb + 32);           // overlaps compute below

        v16h af[2], bf[4];
#pragma unroll
        for (int mt = 0; mt < 2; ++mt)
            af[mt] = load_frag(&As[buf][0][0], wm * 32 + mt * 16, GB_LDA, 0, lane);
#pragma unroll
        for (int nt = 0; nt < 4; ++nt)
            bf[nt] = load_frag(&Bs[buf][0][0], wn * 64 + nt * 16, GB_LDA, 0, lane);
#pragma unroll
        for (int mt = 0; mt < 2; ++mt)
#pragma unroll
            for (int nt = 0; nt < 4; ++nt)
                acc[mt][nt] = wmma16(af[mt], bf[nt], acc[mt][nt]);

        if (hasNext) lstore(buf ^ 1);          // other buffer: no race with readers
        __syncthreads();
    }

    const int h  = lane >> 4;
    const int nl = lane & 15;
#pragma unroll
    for (int mt = 0; mt < 2; ++mt)
#pragma unroll
        for (int nt = 0; nt < 4; ++nt)
#pragma unroll
            for (int v = 0; v < 8; ++v) {
                int row = r0 + wm * 32 + mt * 16 + v + 8 * h;
                int col = c0 + wn * 64 + nt * 16 + nl;
                float val = acc[mt][nt][v] + bias[col];
                if (HEADSPLIT) {
                    int bi = row / S, si = row - bi * S;
                    int hd = col >> 6, dd = col & 63;
                    size_t o = ((((size_t)bi * (N >> 6) + hd) * S + si) << 6) + dd;
                    out[o] = (OT)val;
                } else {
                    out[(size_t)row * N + col] = (OT)val;
                }
            }
}

// ---------- Flash attention ----------
// grid: (S/128, B*NH). 256 threads = 8 waves; wave w owns 16 query rows.
// K tiles: double-buffered LDS, TDM(tensor_load_to_lds) for tile i+1 issued
// before compute of tile i (DMA overlaps compute; s_wait_tensorcnt at the
// staging point). V tiles: global loads for i+1 register-pipelined over
// compute of i, committed to LDS after the read-done barrier.
__global__ __launch_bounds__(256) void attn_kernel(
    const _Float16* __restrict__ Qh, const _Float16* __restrict__ Kh,
    const _Float16* __restrict__ Vh, const unsigned char* __restrict__ mask,
    _Float16* __restrict__ ctx, int B, int NH, int S)
{
    __shared__ alignas(16) _Float16 Ks[2][64][72];    // K tiles [key][d]
    __shared__ alignas(16) _Float16 Vt[64][72];       // V tile transposed [d][key]
    __shared__ alignas(16) _Float16 Ps[8][16][72];    // per-wave P (16 x 64 keys)
    __shared__ float mb[64];

    const int tid  = threadIdx.x;
    const int lane = tid & 31;
    const int w    = tid >> 5;
    const int h    = lane >> 4;
    const int nl   = lane & 15;

    const int bh = blockIdx.y;
    const int bi = bh / NH;
    const int hd = bh - bi * NH;
    const int q0 = blockIdx.x * 128 + w * 16;

    const _Float16* qb = Qh + ((size_t)bi * NH + hd) * S * 64;
    const _Float16* kb = Kh + ((size_t)bi * NH + hd) * S * 64;
    const _Float16* vb = Vh + ((size_t)bi * NH + hd) * S * 64;

    // Q fragments with folded softmax scale (1/8, exact in f16)
    v16h qf[2];
    qf[0] = load_frag(qb, q0, 64, 0,  lane);
    qf[1] = load_frag(qb, q0, 64, 32, lane);
#pragma unroll
    for (int i = 0; i < 16; ++i) {
        qf[0][i] = qf[0][i] * (_Float16)0.125f;
        qf[1][i] = qf[1][i] * (_Float16)0.125f;
    }

    // V staging coordinates + register pipeline
    int vKey[4], vDq[4];
#pragma unroll
    for (int t = 0; t < 4; ++t) {
        int idx = tid + t * 256;
        vKey[t] = idx >> 4;
        vDq[t]  = (idx & 15) << 2;
    }
    v4h vReg[4];
    auto loadV = [&](int kt) {
#pragma unroll
        for (int t = 0; t < 4; ++t)
            vReg[t] = *(const v4h*)(vb + (size_t)(kt + vKey[t]) * 64 + vDq[t]);
    };
    auto storeVt = [&]() {
#pragma unroll
        for (int t = 0; t < 4; ++t) {
            Vt[vDq[t] + 0][vKey[t]] = vReg[t][0];
            Vt[vDq[t] + 1][vKey[t]] = vReg[t][1];
            Vt[vDq[t] + 2][vKey[t]] = vReg[t][2];
            Vt[vDq[t] + 3][vKey[t]] = vReg[t][3];
        }
    };

    v8f acc[4];
#pragma unroll
    for (int dt = 0; dt < 4; ++dt) acc[dt] = zf();
    float m[8], l[8];
#pragma unroll
    for (int v = 0; v < 8; ++v) { m[v] = -1e30f; l[v] = 0.0f; }

    const unsigned int ks_lds0 = (unsigned int)(uintptr_t)&Ks[0][0][0];
    const unsigned int ks_lds1 = (unsigned int)(uintptr_t)&Ks[1][0][0];

    const int NT = S >> 6;   // 64-key tiles

    // ---- prologue: stage tile 0 ----
    if (w == 0) tdm_load_k64x64(ks_lds0, kb);
    loadV(0);
    storeVt();
    if (tid < 64) mb[tid] = mask[(size_t)bi * S + tid] ? 0.0f : -1e30f;
    if (w == 0) wait_tensorcnt0();
    __syncthreads();

    for (int it = 0; it < NT; ++it) {
        const int  kt      = it << 6;
        const bool hasNext = (it + 1 < NT);
        const int  cur     = it & 1;

        // issue DMA for next K tile into the other buffer (overlaps compute)
        if (w == 0 && hasNext)
            tdm_load_k64x64(cur ? ks_lds0 : ks_lds1, kb + ((size_t)kt + 64) * 64);
        // hoist next V tile's global loads over compute
        if (hasNext) loadV(kt + 64);

        // ---- S = (Q*scale) K^T + mask_bias ----
        const _Float16* ksb = &Ks[cur][0][0];
        v8f sc[4];
#pragma unroll
        for (int nt = 0; nt < 4; ++nt) {
            v16h b0 = load_frag(ksb, nt * 16, 72, 0,  lane);
            v16h b1 = load_frag(ksb, nt * 16, 72, 32, lane);
            v8f a = zf();
            a = wmma16(qf[0], b0, a);
            a = wmma16(qf[1], b1, a);
            float mbv = mb[nt * 16 + nl];
#pragma unroll
            for (int v = 0; v < 8; ++v) sc[nt][v] = a[v] + mbv;
        }

        // ---- online softmax (DPP butterflies) ----
        float mn[8], al[8];
#pragma unroll
        for (int v = 0; v < 8; ++v) {
            float x = fmaxf(fmaxf(sc[0][v], sc[1][v]), fmaxf(sc[2][v], sc[3][v]));
            x = red_max16(x);
            mn[v] = fmaxf(m[v], x);
            al[v] = __expf(m[v] - mn[v]);
        }
        float rs[8];
#pragma unroll
        for (int v = 0; v < 8; ++v) rs[v] = 0.0f;
#pragma unroll
        for (int nt = 0; nt < 4; ++nt)
#pragma unroll
            for (int v = 0; v < 8; ++v) {
                float p = __expf(sc[nt][v] - mn[v]);
                sc[nt][v] = p;
                rs[v] += p;
            }
#pragma unroll
        for (int v = 0; v < 8; ++v) {
            l[v] = l[v] * al[v] + red_add16(rs[v]);
            m[v] = mn[v];
        }
#pragma unroll
        for (int dt = 0; dt < 4; ++dt)
#pragma unroll
            for (int v = 0; v < 8; ++v) acc[dt][v] *= al[v];

        // ---- P: C-layout -> A-fragment via per-wave LDS slab ----
#pragma unroll
        for (int nt = 0; nt < 4; ++nt)
#pragma unroll
            for (int v = 0; v < 8; ++v)
                Ps[w][v + 8 * h][nt * 16 + nl] = (_Float16)sc[nt][v];

        // ---- O += P * V ----
#pragma unroll
        for (int kk = 0; kk < 2; ++kk) {
            v16h ap = load_frag(&Ps[w][0][0], 0, 72, kk * 32, lane);
#pragma unroll
            for (int dt = 0; dt < 4; ++dt) {
                v16h bv = load_frag(&Vt[0][0], dt * 16, 72, kk * 32, lane);
                acc[dt] = wmma16(ap, bv, acc[dt]);
            }
        }
        __syncthreads();            // all waves done reading Vt / Ks[cur] / mb

        if (hasNext) {
            storeVt();              // commit pipelined V tile
            if (tid < 64) mb[tid] = mask[(size_t)bi * S + kt + 64 + tid] ? 0.0f : -1e30f;
            if (w == 0) wait_tensorcnt0();   // next K tile DMA complete
            __syncthreads();        // staged tile visible to all waves
        }
    }

    // ---- finalize: ctx[b][s][head*64+d] (f16) ----
#pragma unroll
    for (int dt = 0; dt < 4; ++dt) {
#pragma unroll
        for (int v = 0; v < 8; ++v) {
            int si = q0 + v + 8 * h;
            int dd = dt * 16 + nl;
            float inv = 1.0f / l[v];
            ctx[((size_t)bi * S + si) * (NH << 6) + (hd << 6) + dd] = (_Float16)(acc[dt][v] * inv);
        }
    }
}

// ---------- launcher ----------
extern "C" void kernel_launch(void* const* d_in, const int* in_sizes, int n_in,
                              void* d_out, int out_size, void* d_ws, size_t ws_size,
                              hipStream_t stream) {
    (void)in_sizes; (void)n_in; (void)out_size; (void)ws_size;
    const float* q  = (const float*)d_in[0];
    const float* k  = (const float*)d_in[1];
    const float* v  = (const float*)d_in[2];
    const unsigned char* msk = (const unsigned char*)d_in[3];
    const float* Wq = (const float*)d_in[4];
    const float* bq = (const float*)d_in[5];
    const float* Wk = (const float*)d_in[6];
    const float* bk = (const float*)d_in[7];
    const float* Wv = (const float*)d_in[8];
    const float* bv = (const float*)d_in[9];
    const float* Wo = (const float*)d_in[10];
    const float* bo = (const float*)d_in[11];
    float* out = (float*)d_out;

    const int B = 4, S = 2048, MD = 1024, NH = 16;
    const size_t per = (size_t)B * S * MD;   // 8Mi f16 = 16 MB each
    _Float16* Qh  = (_Float16*)d_ws;
    _Float16* Kh  = Qh + per;
    _Float16* Vh  = Kh + per;
    _Float16* ctx = Vh + per;                // total ws use: 64 MB

    dim3 blk(256);
    dim3 gg(MD / 128, (B * S) / 128);
    gemm_bias_kernel<float, true, _Float16><<<gg, blk, 0, stream>>>(q, Wq, bq, Qh, B * S, MD, MD, S);
    gemm_bias_kernel<float, true, _Float16><<<gg, blk, 0, stream>>>(k, Wk, bk, Kh, B * S, MD, MD, S);
    gemm_bias_kernel<float, true, _Float16><<<gg, blk, 0, stream>>>(v, Wv, bv, Vh, B * S, MD, MD, S);
    attn_kernel<<<dim3(S / 128, B * NH), blk, 0, stream>>>(Qh, Kh, Vh, msk, ctx, B, NH, S);
    gemm_bias_kernel<_Float16, false, float><<<gg, blk, 0, stream>>>(ctx, Wo, bo, out, B * S, MD, MD, S);
}
